// DPN_8864812499591
// MI455X (gfx1250) — matrix operand order repaired and checked by reading, not compiled, not run on hardware
//
#include <hip/hip_runtime.h>
#include <hip/hip_bf16.h>
#include <math.h>

// ---------------------------------------------------------------------------
// Problem constants (from the reference)
// ---------------------------------------------------------------------------
static constexpr int Bb = 2, Gg = 8, Dd = 128, Hh = 80, Ww = 160;
static constexpr int Nn = Bb * Hh * Ww;        // 25600 pixels
static constexpr int NWIN = Bb * Hh * (Ww / 4);// 6400 stripe windows of 16 tokens
static constexpr float EPSV = 1e-3f;

typedef __attribute__((ext_vector_type(16))) __bf16 bf16x16;
typedef __attribute__((ext_vector_type(8)))  float  f32x8;

#define WMMA_B16(a, b, c) \
  __builtin_amdgcn_wmma_f32_16x16x32_bf16(false, (a), false, (b), (short)0, (c), false, false)

// ---------------------------------------------------------------------------
// CDNA5 async global->LDS copy (ASYNCcnt-tracked, bypasses VGPRs).
// One 16-byte piece per lane. lds_off is the 32-bit LDS byte offset
// (low 32 bits of the generic address of a __shared__ object).
// ---------------------------------------------------------------------------
__device__ __forceinline__ void lds_async_b128(unsigned lds_off, const void* g) {
  asm volatile("global_load_async_to_lds_b128 %0, %1, off"
               :: "v"(lds_off), "v"((unsigned long long)(uintptr_t)g)
               : "memory");
}
__device__ __forceinline__ void wait_async0() {
  asm volatile("s_wait_asynccnt 0x0" ::: "memory");
}

// Stage 2048 contiguous floats from gsrc into a [16][132]-padded LDS tile.
// 512 16B pieces; 32 pieces per 128-float row so pieces never cross padding.
__device__ __forceinline__ void stage_window_async(float* lds_tile, const float* gsrc, int tid) {
  unsigned base = (unsigned)(uintptr_t)(const void*)lds_tile;
#pragma unroll
  for (int r = 0; r < 4; ++r) {
    int piece = tid + r * 128;
    int row = piece >> 5;
    int col = (piece & 31) * 4;
    lds_async_b128(base + (unsigned)(row * 132 + col) * 4u, gsrc + piece * 4);
  }
  wait_async0();
}

// ---------------------------------------------------------------------------
// WMMA fragment helpers (wave32, 16x16x32 bf16)
// A 16(M)x32(K):  lane<16 -> row M=lane,   K = k0+0..15
//                 lane>=16 -> row M=lane-16, K = k0+16..31
// B 32(K)x16(N):  lane<16 -> col N=lane,   K = 0..15 ; lane>=16 -> K = 16..31
// C/D 16x16 f32:  lane holds col N=lane&15, rows (lane>=16 ? 8..15 : 0..7)
// ---------------------------------------------------------------------------
__device__ __forceinline__ bf16x16 a_frag_f32(const float* m, int ld, int k0, int lane) {
  int r = lane & 15;
  int kb = k0 + ((lane & 16) ? 16 : 0);
  const float* p = m + r * ld + kb;
  bf16x16 a;
#pragma unroll
  for (int i = 0; i < 16; ++i) a[i] = (__bf16)p[i];
  return a;
}

// A 16x16 (K padded to 32 with zeros), for att@V
__device__ __forceinline__ bf16x16 a_frag_pad16(const float* m, int ld, int lane) {
  int r = lane & 15;
  bool hi = (lane & 16) != 0;
  bf16x16 a;
#pragma unroll
  for (int i = 0; i < 16; ++i) a[i] = hi ? (__bf16)0.f : (__bf16)m[r * ld + i];
  return a;
}

// B fragment from pre-packed weight tiles: 512 bf16 per 32x16 tile,
// lane-major so each lane reads 16 contiguous bf16 (32 bytes).
__device__ __forceinline__ bf16x16 b_frag_packed(const __bf16* tile, int lane) {
  return *(const bf16x16*)(tile + lane * 16);
}

// B[k][n] = src[n*ld + koff + k]   (rows of src are the N dimension) -> QK^T
__device__ __forceinline__ bf16x16 b_frag_transpose(const float* src, int ld, int koff, int lane) {
  int n = lane & 15;
  int kb = (lane & 16) ? 16 : 0;
  bf16x16 b;
#pragma unroll
  for (int i = 0; i < 16; ++i) b[i] = (__bf16)src[n * ld + koff + kb + i];
  return b;
}

// B[k][n] = src[k*ld + coff + n], valid K = 0..15, K=16..31 zero-padded -> att@V
__device__ __forceinline__ bf16x16 b_frag_rows_pad16(const float* src, int ld, int coff, int lane) {
  int n = lane & 15;
  bf16x16 b;
  if (lane < 16) {
#pragma unroll
    for (int i = 0; i < 16; ++i) b[i] = (__bf16)src[i * ld + coff + n];
  } else {
#pragma unroll
    for (int i = 0; i < 16; ++i) b[i] = (__bf16)0.f;
  }
  return b;
}

__device__ __forceinline__ void store_d_f32(float* m, int ld, f32x8 d, int lane) {
  int c = lane & 15;
  int rb = (lane >> 4) * 8;
#pragma unroll
  for (int i = 0; i < 8; ++i) m[(rb + i) * ld + c] = d[i];
}

// ---------------------------------------------------------------------------
// Weight packer: f32 -> bf16 WMMA-B tile layout.
// mode 0: src row-major [K][N]
// mode 1: conv3x3 proj_w1 [oc=n][ic][3][3], k = tap*128 + ic
// mode 2: combined embed(72) + zeros(24) + ctx(64) rows, N=128
// mode 3: src [N][K] (transpose)
// ---------------------------------------------------------------------------
__global__ void k_pack(const float* __restrict__ s0, const float* __restrict__ s1,
                       __bf16* __restrict__ dst, int K, int N, int mode) {
  int idx = blockIdx.x * blockDim.x + threadIdx.x;
  if (idx >= K * N) return;
  int k = idx / N, n = idx % N;
  float v;
  if (mode == 0) {
    v = s0[(size_t)k * N + n];
  } else if (mode == 1) {
    int tap = k >> 7, ic = k & 127;
    v = s0[(((size_t)n * 128 + ic) * 3 + tap / 3) * 3 + (tap % 3)];
  } else if (mode == 2) {
    if (k < 72)      v = s0[(size_t)k * N + n];
    else if (k < 96) v = 0.f;
    else             v = s1[(size_t)(k - 96) * N + n];
  } else {
    v = s0[(size_t)n * K + k];
  }
  int kt = k >> 5, k0 = k & 31, nt = n >> 4, n0 = n & 15;
  int lane = n0 + ((k0 & 16) ? 16 : 0);
  int half = k0 & 15;
  dst[(((size_t)kt * (N >> 4) + nt) * 32 + lane) * 16 + half] = (__bf16)v;
}

__global__ void k_zero(float* p, int n) {
  int i = blockIdx.x * blockDim.x + threadIdx.x;
  if (i < n) p[i] = 0.f;
}

// ---------------------------------------------------------------------------
// cost_volume [B,G,D,H,W] -> cv [N,G,D]  (first tuple output; feeds the filter)
// ---------------------------------------------------------------------------
__global__ void k_transpose_cv(const float* __restrict__ src, float* __restrict__ dst) {
  long idx = (long)blockIdx.x * blockDim.x + threadIdx.x;
  const long total = (long)Bb * Gg * Dd * Hh * Ww;
  if (idx >= total) return;
  int w = (int)(idx % Ww); long t = idx / Ww;
  int h = (int)(t % Hh); t /= Hh;
  int d = (int)(t % Dd); t /= Dd;
  int g = (int)(t % Gg);
  int b = (int)(t / Gg);
  int n = (b * Hh + h) * Ww + w;
  dst[((size_t)n * Gg + g) * Dd + d] = src[idx];
}

// ---------------------------------------------------------------------------
// Per-pixel cost filter (conv1d G->8->16->1) + softmax + local-max top-4 seeds.
// One 128-thread workgroup per pixel; thread = disparity.
// ---------------------------------------------------------------------------
__global__ __launch_bounds__(128) void k_costfilter(
    const float* __restrict__ cv,
    const float* __restrict__ w1, const float* __restrict__ b1,
    const float* __restrict__ w2, const float* __restrict__ b2,
    const float* __restrict__ w3, const float* __restrict__ b3,
    float* __restrict__ prob_out, float* __restrict__ seeds_out) {
  __shared__ float s_cv[8][128];
  __shared__ float s_h1[8][128];
  __shared__ float s_h2[16][128];
  __shared__ float s_w[1072];
  __shared__ float s_red[128];
  __shared__ int   s_ri[128];
  const int d = threadIdx.x;
  const int n = blockIdx.x;

  for (int i = d; i < 320; i += 128) s_w[i] = w1[i];
  if (d < 8)  s_w[320 + d] = b1[d];
  for (int i = d; i < 640; i += 128) s_w[328 + i] = w2[i];
  if (d < 16) s_w[968 + d] = b2[d];
  if (d < 80) s_w[984 + d] = w3[d];
  if (d == 0) s_w[1064] = b3[0];
#pragma unroll
  for (int g = 0; g < 8; ++g) s_cv[g][d] = cv[((size_t)n * 8 + g) * 128 + d];
  __syncthreads();

  // conv1: 8 in -> 8 out, k=5, zero pad
#pragma unroll
  for (int co = 0; co < 8; ++co) {
    float acc = s_w[320 + co];
#pragma unroll
    for (int ci = 0; ci < 8; ++ci)
#pragma unroll
      for (int t = 0; t < 5; ++t) {
        int dd = d + t - 2;
        if (dd >= 0 && dd < 128) acc += s_w[(co * 8 + ci) * 5 + t] * s_cv[ci][dd];
      }
    s_h1[co][d] = fmaxf(acc, 0.f);
  }
  __syncthreads();
  // conv2: 8 -> 16
#pragma unroll
  for (int co = 0; co < 16; ++co) {
    float acc = s_w[968 + co];
#pragma unroll
    for (int ci = 0; ci < 8; ++ci)
#pragma unroll
      for (int t = 0; t < 5; ++t) {
        int dd = d + t - 2;
        if (dd >= 0 && dd < 128) acc += s_w[328 + (co * 8 + ci) * 5 + t] * s_h1[ci][dd];
      }
    s_h2[co][d] = fmaxf(acc, 0.f);
  }
  __syncthreads();
  // conv3: 16 -> 1
  float cost = s_w[1064];
#pragma unroll
  for (int ci = 0; ci < 16; ++ci)
#pragma unroll
    for (int t = 0; t < 5; ++t) {
      int dd = d + t - 2;
      if (dd >= 0 && dd < 128) cost += s_w[984 + ci * 5 + t] * s_h2[ci][dd];
    }

  // softmax over disparity
  s_red[d] = cost; __syncthreads();
  for (int s = 64; s > 0; s >>= 1) { if (d < s) s_red[d] = fmaxf(s_red[d], s_red[d + s]); __syncthreads(); }
  float mx = s_red[0]; __syncthreads();
  float e = expf(cost - mx);
  s_red[d] = e; __syncthreads();
  for (int s = 64; s > 0; s >>= 1) { if (d < s) s_red[d] += s_red[d + s]; __syncthreads(); }
  float p = e / s_red[0];
  __syncthreads();
  prob_out[(size_t)n * 128 + d] = p;

  // local-max suppression
  s_red[d] = p; __syncthreads();
  float pm1 = d > 0   ? s_red[d - 1] : -INFINITY;
  float pp1 = d < 127 ? s_red[d + 1] : -INFINITY;
  float pooled = fmaxf(fmaxf(pm1, p), pp1);
  bool nlm = (p != pooled) && (p > EPSV);
  float pv = nlm ? EPSV : p;
  __syncthreads();

  // top-4 (stable: ties -> lower index)
  for (int it = 0; it < 4; ++it) {
    s_red[d] = pv; s_ri[d] = d; __syncthreads();
    for (int s = 64; s > 0; s >>= 1) {
      if (d < s) {
        float v2 = s_red[d + s]; int i2 = s_ri[d + s];
        if (v2 > s_red[d] || (v2 == s_red[d] && i2 < s_ri[d])) { s_red[d] = v2; s_ri[d] = i2; }
      }
      __syncthreads();
    }
    int best = s_ri[0];
    if (d == 0) seeds_out[(size_t)n * 4 + it] = (float)best;   // numeric index
    if (d == best) pv = -INFINITY;
    __syncthreads();
  }
}

// ---------------------------------------------------------------------------
// 3x3 conv (128->128, pad 1) as implicit GEMM on WMMA. 256 threads = 8 waves,
// 16-pixel row strip per block; each wave owns one 16-channel output tile.
// Also accumulates per-(b,channel) sums for instance-norm via f32 atomics.
// ---------------------------------------------------------------------------
__global__ __launch_bounds__(256) void k_conv3x3(
    const float* __restrict__ fmap, const __bf16* __restrict__ pw1,
    float* __restrict__ convout, float* __restrict__ sum, float* __restrict__ sumsq) {
  __shared__ float sA[16][132];
  const int tid = threadIdx.x, lane = tid & 31, wv = tid >> 5;
  int blk = blockIdx.x;                        // B*H*(W/16)
  int wt = blk % (Ww / 16); int t2 = blk / (Ww / 16);
  int h = t2 % Hh; int b = t2 / Hh;
  int w0 = wt * 16;
  __builtin_prefetch(pw1 + (size_t)wv * 512, 0, 3);

  f32x8 acc = {};
  for (int tap = 0; tap < 9; ++tap) {
    int ty = tap / 3 - 1, tx = tap % 3 - 1;
    int hh = h + ty;
    __syncthreads();
    for (int e2 = tid; e2 < 16 * 128; e2 += 256) {
      int px = e2 >> 7, ic = e2 & 127;
      int ww = w0 + px + tx;
      float v = 0.f;
      if (hh >= 0 && hh < Hh && ww >= 0 && ww < Ww)
        v = fmap[(((size_t)b * 128 + ic) * Hh + hh) * Ww + ww];
      sA[px][ic] = v;
    }
    __syncthreads();
#pragma unroll
    for (int kt = 0; kt < 4; ++kt) {
      bf16x16 a = a_frag_f32(&sA[0][0], 132, kt * 32, lane);
      bf16x16 bb = b_frag_packed(pw1 + ((size_t)((tap * 4 + kt) * 8 + wv)) * 512, lane);
      acc = WMMA_B16(a, bb, acc);
    }
  }
  int c = lane & 15, rb = (lane >> 4) * 8;
  int oc = wv * 16 + c;
  float s = 0.f, sq = 0.f;
#pragma unroll
  for (int i = 0; i < 8; ++i) {
    int n = (b * Hh + h) * Ww + (w0 + rb + i);
    float v = acc[i];
    convout[(size_t)n * 128 + oc] = v;
    s += v; sq += v * v;
  }
  atomicAdd(&sum[b * 128 + oc], s);
  atomicAdd(&sumsq[b * 128 + oc], sq);
}

__global__ void k_in_finalize(const float* __restrict__ sum, const float* __restrict__ sumsq,
                              float* __restrict__ mu, float* __restrict__ rsig) {
  int i = threadIdx.x;                          // 256 = B*128
  const float inv = 1.0f / (float)(Hh * Ww);
  float m = sum[i] * inv;
  float v = sumsq[i] * inv - m * m;
  mu[i] = m;
  rsig[i] = rsqrtf(v + 1e-5f);
}

// ---------------------------------------------------------------------------
// instance-norm + ReLU + 1x1 conv (128->64) on WMMA. 4 waves / 16 pixels.
// ---------------------------------------------------------------------------
__global__ __launch_bounds__(128) void k_ctx(
    const float* __restrict__ convout, const float* __restrict__ mu,
    const float* __restrict__ rsig, const __bf16* __restrict__ pw2,
    float* __restrict__ ctx) {
  __shared__ float sA[16][132];
  const int tid = threadIdx.x, lane = tid & 31, wv = tid >> 5;
  int n0 = blockIdx.x * 16;
  int b = n0 / (Hh * Ww);
  for (int e = tid; e < 16 * 128; e += 128) {
    int px = e >> 7, ic = e & 127;
    float v = (convout[(size_t)(n0 + px) * 128 + ic] - mu[b * 128 + ic]) * rsig[b * 128 + ic];
    sA[px][ic] = fmaxf(v, 0.f);
  }
  __syncthreads();
  f32x8 acc = {};
#pragma unroll
  for (int kt = 0; kt < 4; ++kt) {
    bf16x16 a = a_frag_f32(&sA[0][0], 132, kt * 32, lane);
    bf16x16 bb = b_frag_packed(pw2 + ((size_t)(kt * 4 + wv)) * 512, lane);
    acc = WMMA_B16(a, bb, acc);
  }
  int c = lane & 15, rb = (lane >> 4) * 8;
#pragma unroll
  for (int i = 0; i < 8; ++i)
    ctx[(size_t)(n0 + rb + i) * 64 + wv * 16 + c] = acc[i];
}

// ---------------------------------------------------------------------------
// Seed embedding: gather +-4 cost window (72 vals) || zeros(24) || ctx(64)
// then one combined K=160 WMMA GEMM -> x tokens, written in window layout.
// Block = 4 pixels x 4 proposals = one 16-token window. 2 N-tiles per wave,
// kt-outer so each A fragment is built once.
// ---------------------------------------------------------------------------
__global__ __launch_bounds__(128) void k_embed(
    const float* __restrict__ cv, const float* __restrict__ seeds_f,
    const float* __restrict__ ctx, const __bf16* __restrict__ pemb,
    const float* __restrict__ embed_b, const float* __restrict__ ctx_b,
    float* __restrict__ xbuf) {
  __shared__ float sA[16][164];
  const int tid = threadIdx.x, lane = tid & 31, wv = tid >> 5;
  int n0 = blockIdx.x * 4;
  for (int e = tid; e < 16 * 160; e += 128) {
    int t = e / 160, k = e % 160;
    int px = t >> 2, p = t & 3;
    int n = n0 + px;
    float v;
    if (k < 72) {
      int g = k / 9, tt = k % 9;
      int sd = (int)seeds_f[(size_t)n * 4 + p];
      int dd = sd + tt - 4;
      dd = dd < 0 ? 0 : (dd > 127 ? 127 : dd);
      v = cv[((size_t)n * 8 + g) * 128 + dd];
    } else if (k < 96) {
      v = 0.f;
    } else {
      v = ctx[(size_t)n * 64 + (k - 96)];
    }
    sA[t][k] = v;
  }
  __syncthreads();
  f32x8 acc[2] = {};
  for (int kt = 0; kt < 5; ++kt) {
    bf16x16 a = a_frag_f32(&sA[0][0], 164, kt * 32, lane);
#pragma unroll
    for (int t = 0; t < 2; ++t) {
      int j = wv + t * 4;
      bf16x16 bb = b_frag_packed(pemb + ((size_t)(kt * 8 + j)) * 512, lane);
      acc[t] = WMMA_B16(a, bb, acc[t]);
    }
  }
  int c = lane & 15, rb = (lane >> 4) * 8;
#pragma unroll
  for (int t = 0; t < 2; ++t) {
    int col = (wv + t * 4) * 16 + c;
    float bias = embed_b[col] + ctx_b[col];
#pragma unroll
    for (int i = 0; i < 8; ++i)
      xbuf[((size_t)blockIdx.x * 16 + rb + i) * 128 + col] = acc[t][i] + bias;
  }
}

// ---------------------------------------------------------------------------
// One transformer layer for one 16-token window. 128 threads = 4 waves
// (one attention head per wave). All GEMMs on WMMA bf16, f32 accumulation.
// kt-outer / multi-accumulator so A fragments are built once per K step.
// ---------------------------------------------------------------------------
__global__ __launch_bounds__(128) void k_xformer(
    float* __restrict__ xbuf,
    const __bf16* __restrict__ pqkv, const __bf16* __restrict__ pattn,
    const __bf16* __restrict__ pfc1, const __bf16* __restrict__ pfc2,
    const float* __restrict__ qkvb, const float* __restrict__ attnb,
    const float* __restrict__ g1, const float* __restrict__ b1v,
    const float* __restrict__ fc1b, const float* __restrict__ fc2b,
    const float* __restrict__ g2, const float* __restrict__ b2v) {
  __shared__ float sx[16][132];   // token activations
  __shared__ float sy[16][520];   // qkv (cols 0..383), attn-out (cols 384..511), fc1 (0..511)
  __shared__ float so[16][132];   // attention scores / pre-LN sums
  const int tid = threadIdx.x, lane = tid & 31, wv = tid >> 5;
  float* xg = xbuf + (size_t)blockIdx.x * 16 * 128;
  __builtin_prefetch(pqkv + (size_t)wv * 512, 0, 3);
  __builtin_prefetch(pfc1 + (size_t)wv * 512, 0, 3);

  // async global->LDS window load (ASYNCcnt path)
  stage_window_async(&sx[0][0], xg, tid);
  __syncthreads();

  // ---- qkv: [16x128] @ [128x384], 6 N-tiles per wave ----
  {
    f32x8 acc[6] = {};
    for (int kt = 0; kt < 4; ++kt) {
      bf16x16 a = a_frag_f32(&sx[0][0], 132, kt * 32, lane);
#pragma unroll
      for (int t = 0; t < 6; ++t) {
        int j = wv + t * 4;
        bf16x16 bb = b_frag_packed(pqkv + ((size_t)(kt * 24 + j)) * 512, lane);
        acc[t] = WMMA_B16(a, bb, acc[t]);
      }
    }
    int c = lane & 15, rb = (lane >> 4) * 8;
#pragma unroll
    for (int t = 0; t < 6; ++t) {
      int j = wv + t * 4;
      float bias = qkvb[j * 16 + c];
#pragma unroll
      for (int i = 0; i < 8; ++i) sy[rb + i][j * 16 + c] = acc[t][i] + bias;
    }
  }
  __syncthreads();

  // ---- S = Q K^T * scale (one head per wave) ----
  {
    const int hh = wv;
    const float scale = 0.17677669529663687f;   // 1/sqrt(32)
    bf16x16 aq = a_frag_f32(&sy[0][hh * 32], 520, 0, lane);
    bf16x16 bk = b_frag_transpose(&sy[0][0], 520, 128 + hh * 32, lane);
    f32x8 z = {};
    f32x8 s = WMMA_B16(aq, bk, z);
#pragma unroll
    for (int i = 0; i < 8; ++i) s[i] *= scale;
    store_d_f32(&so[0][hh * 16], 132, s, lane);
  }
  __syncthreads();

  // ---- softmax rows (64 rows total: 4 heads x 16 queries) ----
  if (tid < 64) {
    int hh = tid >> 4, m = tid & 15;
    float* r = &so[m][hh * 16];
    float mx = -INFINITY;
#pragma unroll
    for (int i = 0; i < 16; ++i) mx = fmaxf(mx, r[i]);
    float sm = 0.f;
#pragma unroll
    for (int i = 0; i < 16; ++i) { float e = expf(r[i] - mx); r[i] = e; sm += e; }
    float inv = 1.f / sm;
#pragma unroll
    for (int i = 0; i < 16; ++i) r[i] *= inv;
  }
  __syncthreads();

  // ---- O = att @ V (K padded 16 -> 32) ----
  {
    const int hh = wv;
    bf16x16 aa = a_frag_pad16(&so[0][hh * 16], 132, lane);
#pragma unroll
    for (int nt = 0; nt < 2; ++nt) {
      bf16x16 bv = b_frag_rows_pad16(&sy[0][0], 520, 256 + hh * 32 + nt * 16, lane);
      f32x8 z = {};
      f32x8 o = WMMA_B16(aa, bv, z);
      store_d_f32(&sy[0][384 + hh * 32 + nt * 16], 520, o, lane);
    }
  }
  __syncthreads();

  // ---- out proj + residual (2 N-tiles per wave) ----
  {
    f32x8 acc[2] = {};
    for (int kt = 0; kt < 4; ++kt) {
      bf16x16 a = a_frag_f32(&sy[0][384], 520, kt * 32, lane);
#pragma unroll
      for (int t = 0; t < 2; ++t) {
        bf16x16 bb = b_frag_packed(pattn + ((size_t)(kt * 8 + wv + t * 4)) * 512, lane);
        acc[t] = WMMA_B16(a, bb, acc[t]);
      }
    }
    int c = lane & 15, rb = (lane >> 4) * 8;
#pragma unroll
    for (int t = 0; t < 2; ++t) {
      int col = (wv + t * 4) * 16 + c;
      float bias = attnb[col];
#pragma unroll
      for (int i = 0; i < 8; ++i) so[rb + i][col] = acc[t][i] + bias + sx[rb + i][col];
    }
  }
  __syncthreads();

  // ---- LN1 -> sx ----
  if (tid < 16) {
    float mu = 0.f;
    for (int c = 0; c < 128; ++c) mu += so[tid][c];
    mu *= (1.f / 128.f);
    float var = 0.f;
    for (int c = 0; c < 128; ++c) { float dl = so[tid][c] - mu; var += dl * dl; }
    var *= (1.f / 128.f);
    float inv = rsqrtf(var + 1e-5f);
    for (int c = 0; c < 128; ++c) sx[tid][c] = (so[tid][c] - mu) * inv * g1[c] + b1v[c];
  }
  __syncthreads();

  // ---- fc1 + exact GELU (8 N-tiles per wave) ----
  {
    f32x8 acc[8] = {};
    for (int kt = 0; kt < 4; ++kt) {
      bf16x16 a = a_frag_f32(&sx[0][0], 132, kt * 32, lane);
#pragma unroll
      for (int t = 0; t < 8; ++t) {
        int j = wv + t * 4;
        bf16x16 bb = b_frag_packed(pfc1 + ((size_t)(kt * 32 + j)) * 512, lane);
        acc[t] = WMMA_B16(a, bb, acc[t]);
      }
    }
    int c = lane & 15, rb = (lane >> 4) * 8;
#pragma unroll
    for (int t = 0; t < 8; ++t) {
      int j = wv + t * 4;
      float bias = fc1b[j * 16 + c];
#pragma unroll
      for (int i = 0; i < 8; ++i) {
        float v = acc[t][i] + bias;
        sy[rb + i][j * 16 + c] = 0.5f * v * (1.f + erff(v * 0.7071067811865475f));
      }
    }
  }
  __syncthreads();

  // ---- fc2 + residual (2 N-tiles per wave, K=512) ----
  {
    f32x8 acc[2] = {};
    for (int kt = 0; kt < 16; ++kt) {
      bf16x16 a = a_frag_f32(&sy[0][0], 520, kt * 32, lane);
#pragma unroll
      for (int t = 0; t < 2; ++t) {
        bf16x16 bb = b_frag_packed(pfc2 + ((size_t)(kt * 8 + wv + t * 4)) * 512, lane);
        acc[t] = WMMA_B16(a, bb, acc[t]);
      }
    }
    int c = lane & 15, rb = (lane >> 4) * 8;
#pragma unroll
    for (int t = 0; t < 2; ++t) {
      int col = (wv + t * 4) * 16 + c;
      float bias = fc2b[col];
#pragma unroll
      for (int i = 0; i < 8; ++i) so[rb + i][col] = acc[t][i] + bias + sx[rb + i][col];
    }
  }
  __syncthreads();

  // ---- LN2 -> global x ----
  if (tid < 16) {
    float mu = 0.f;
    for (int c = 0; c < 128; ++c) mu += so[tid][c];
    mu *= (1.f / 128.f);
    float var = 0.f;
    for (int c = 0; c < 128; ++c) { float dl = so[tid][c] - mu; var += dl * dl; }
    var *= (1.f / 128.f);
    float inv = rsqrtf(var + 1e-5f);
    for (int c = 0; c < 128; ++c) xg[tid * 128 + c] = (so[tid][c] - mu) * inv * g2[c] + b2v[c];
  }
}

// ---------------------------------------------------------------------------
// Final norm + 3-layer head (WMMA for h1/h2, dot for h3) -> labels.
// ---------------------------------------------------------------------------
__global__ __launch_bounds__(128) void k_head(
    const float* __restrict__ xbuf, const float* __restrict__ seeds_f,
    const float* __restrict__ nfg, const float* __restrict__ nfb,
    const __bf16* __restrict__ ph1, const float* __restrict__ h1b,
    const __bf16* __restrict__ ph2, const float* __restrict__ h2b,
    const float* __restrict__ h3w, const float* __restrict__ h3b,
    float* __restrict__ labels) {
  __shared__ float sm[16][132];
  __shared__ float s1[16][132];
  __shared__ float s2[16][132];
  const int tid = threadIdx.x, lane = tid & 31, wv = tid >> 5;
  const float* xg = xbuf + (size_t)blockIdx.x * 2048;
  __builtin_prefetch(ph1 + (size_t)wv * 512, 0, 3);

  stage_window_async(&s2[0][0], xg, tid);      // async global->LDS
  __syncthreads();
  if (tid < 16) {
    float mu = 0.f;
    for (int c = 0; c < 128; ++c) mu += s2[tid][c];
    mu *= (1.f / 128.f);
    float var = 0.f;
    for (int c = 0; c < 128; ++c) { float dl = s2[tid][c] - mu; var += dl * dl; }
    var *= (1.f / 128.f);
    float inv = rsqrtf(var + 1e-5f);
    for (int c = 0; c < 128; ++c) sm[tid][c] = (s2[tid][c] - mu) * inv * nfg[c] + nfb[c];
  }
  __syncthreads();
  {                                             // h1 + relu
    f32x8 acc[2] = {};
    for (int kt = 0; kt < 4; ++kt) {
      bf16x16 a = a_frag_f32(&sm[0][0], 132, kt * 32, lane);
#pragma unroll
      for (int t = 0; t < 2; ++t) {
        bf16x16 bb = b_frag_packed(ph1 + ((size_t)(kt * 8 + wv + t * 4)) * 512, lane);
        acc[t] = WMMA_B16(a, bb, acc[t]);
      }
    }
    int c = lane & 15, rb = (lane >> 4) * 8;
#pragma unroll
    for (int t = 0; t < 2; ++t) {
      int col = (wv + t * 4) * 16 + c;
      float bias = h1b[col];
#pragma unroll
      for (int i = 0; i < 8; ++i) s1[rb + i][col] = fmaxf(acc[t][i] + bias, 0.f);
    }
  }
  __syncthreads();
  {                                             // h2 + relu
    f32x8 acc[2] = {};
    for (int kt = 0; kt < 4; ++kt) {
      bf16x16 a = a_frag_f32(&s1[0][0], 132, kt * 32, lane);
#pragma unroll
      for (int t = 0; t < 2; ++t) {
        bf16x16 bb = b_frag_packed(ph2 + ((size_t)(kt * 8 + wv + t * 4)) * 512, lane);
        acc[t] = WMMA_B16(a, bb, acc[t]);
      }
    }
    int c = lane & 15, rb = (lane >> 4) * 8;
#pragma unroll
    for (int t = 0; t < 2; ++t) {
      int col = (wv + t * 4) * 16 + c;
      float bias = h2b[col];
#pragma unroll
      for (int i = 0; i < 8; ++i) s2[rb + i][col] = fmaxf(acc[t][i] + bias, 0.f);
    }
  }
  __syncthreads();
  if (tid < 16) {                               // h3 dot + labels
    float o = h3b[0];
    for (int c = 0; c < 128; ++c) o += s2[tid][c] * h3w[c];
    int px = tid >> 2, p = tid & 3;
    size_t n = (size_t)blockIdx.x * 4 + px;
    labels[n * 4 + p] = fmaxf(o + seeds_f[n * 4 + p], 0.f);
  }
}

// ---------------------------------------------------------------------------
// Host orchestration
// ---------------------------------------------------------------------------
extern "C" void kernel_launch(void* const* d_in, const int* in_sizes, int n_in,
                              void* d_out, int out_size, void* d_ws, size_t ws_size,
                              hipStream_t stream) {
  (void)in_sizes; (void)n_in; (void)out_size; (void)ws_size;
  const float* cost_volume = (const float*)d_in[0];
  const float* fmap1   = (const float*)d_in[1];
  const float* mlp_w1  = (const float*)d_in[2];
  const float* mlp_b1  = (const float*)d_in[3];
  const float* mlp_w2  = (const float*)d_in[4];
  const float* mlp_b2  = (const float*)d_in[5];
  const float* mlp_w3  = (const float*)d_in[6];
  const float* mlp_b3  = (const float*)d_in[7];
  const float* proj_w1 = (const float*)d_in[8];
  const float* proj_w2 = (const float*)d_in[9];
  const float* embed_w = (const float*)d_in[10];
  const float* embed_b = (const float*)d_in[11];
  const float* ctx_w   = (const float*)d_in[12];
  const float* ctx_b   = (const float*)d_in[13];
  const float* ln1_g   = (const float*)d_in[14];
  const float* ln1_b   = (const float*)d_in[15];
  const float* qkv_w   = (const float*)d_in[16];
  const float* qkv_b   = (const float*)d_in[17];
  const float* attn_w  = (const float*)d_in[18];
  const float* attn_b  = (const float*)d_in[19];
  const float* ln2_g   = (const float*)d_in[20];
  const float* ln2_b   = (const float*)d_in[21];
  const float* fc1_w   = (const float*)d_in[22];
  const float* fc1_b   = (const float*)d_in[23];
  const float* fc2_w   = (const float*)d_in[24];
  const float* fc2_b   = (const float*)d_in[25];
  const float* normf_g = (const float*)d_in[26];
  const float* normf_b = (const float*)d_in[27];
  const float* h1_w    = (const float*)d_in[28];
  const float* h1_b    = (const float*)d_in[29];
  const float* h2_w    = (const float*)d_in[30];
  const float* h2_b    = (const float*)d_in[31];
  const float* h3_w    = (const float*)d_in[32];
  const float* h3_b    = (const float*)d_in[33];

  float* out       = (float*)d_out;
  float* out_cv    = out;                       // [N,G,D]      26,214,400
  float* out_prob  = out + 26214400;            // [N,D]         3,276,800
  float* out_seeds = out_prob + 3276800;        // [N,P]           102,400 (indices as float)
  float* out_labels= out_seeds + 102400;        // [1,N,P]         102,400

  char* ws = (char*)d_ws;
  __bf16* packed = (__bf16*)ws;                 // 602,112 bf16 = 1,204,224 B
  __bf16* p_qkv  = packed + 0;                  // 2 x 49152
  __bf16* p_attn = packed + 98304;              // 2 x 16384
  __bf16* p_fc1  = packed + 131072;             // 2 x 65536
  __bf16* p_fc2  = packed + 262144;             // 2 x 65536
  __bf16* p_h1   = packed + 393216;             // 16384
  __bf16* p_h2   = packed + 409600;             // 16384
  __bf16* p_emb  = packed + 425984;             // 20480
  __bf16* p_w1   = packed + 446464;             // 147456
  __bf16* p_w2   = packed + 593920;             // 8192
  float* stats   = (float*)(ws + 1204224);      // sum|sumsq|mu|rsig (4x256)
  float* ssum    = stats;
  float* ssq     = stats + 256;
  float* smu     = stats + 512;
  float* srs     = stats + 768;
  float* convout = (float*)(ws + 1208320);      // [N,128] f32   13.1 MB
  float* ctxb    = (float*)(ws + 14315520);     // [N,64]  f32    6.6 MB
  float* xbuf    = (float*)(ws + 20869120);     // [6400*16,128] 52.4 MB

  auto pack = [&](const float* s0, const float* s1, __bf16* dst, int K, int N, int mode) {
    int tot = K * N;
    k_pack<<<(tot + 255) / 256, 256, 0, stream>>>(s0, s1, dst, K, N, mode);
  };
  for (int l = 0; l < 2; ++l) {
    pack(qkv_w + (size_t)l * 128 * 384, nullptr, p_qkv + (size_t)l * 49152, 128, 384, 0);
    pack(attn_w + (size_t)l * 128 * 128, nullptr, p_attn + (size_t)l * 16384, 128, 128, 0);
    pack(fc1_w + (size_t)l * 128 * 512, nullptr, p_fc1 + (size_t)l * 65536, 128, 512, 0);
    pack(fc2_w + (size_t)l * 512 * 128, nullptr, p_fc2 + (size_t)l * 65536, 512, 128, 0);
  }
  pack(h1_w, nullptr, p_h1, 128, 128, 0);
  pack(h2_w, nullptr, p_h2, 128, 128, 0);
  pack(embed_w, ctx_w, p_emb, 160, 128, 2);
  pack(proj_w1, nullptr, p_w1, 1152, 128, 1);
  pack(proj_w2, nullptr, p_w2, 128, 64, 3);

  k_zero<<<2, 256, 0, stream>>>(ssum, 512);
  k_transpose_cv<<<102400, 256, 0, stream>>>(cost_volume, out_cv);
  k_costfilter<<<Nn, 128, 0, stream>>>(out_cv, mlp_w1, mlp_b1, mlp_w2, mlp_b2,
                                       mlp_w3, mlp_b3, out_prob, out_seeds);
  k_conv3x3<<<Bb * Hh * (Ww / 16), 256, 0, stream>>>(fmap1, p_w1, convout, ssum, ssq);
  k_in_finalize<<<1, 256, 0, stream>>>(ssum, ssq, smu, srs);
  k_ctx<<<Nn / 16, 128, 0, stream>>>(convout, smu, srs, p_w2, ctxb);
  k_embed<<<NWIN, 128, 0, stream>>>(out_cv, out_seeds, ctxb, p_emb, embed_b, ctx_b, xbuf);
  for (int l = 0; l < 2; ++l) {
    k_xformer<<<NWIN, 128, 0, stream>>>(xbuf,
        p_qkv + (size_t)l * 49152, p_attn + (size_t)l * 16384,
        p_fc1 + (size_t)l * 65536, p_fc2 + (size_t)l * 65536,
        qkv_b + (size_t)l * 384, attn_b + (size_t)l * 128,
        ln1_g + (size_t)l * 128, ln1_b + (size_t)l * 128,
        fc1_b + (size_t)l * 512, fc2_b + (size_t)l * 128,
        ln2_g + (size_t)l * 128, ln2_b + (size_t)l * 128);
  }
  k_head<<<NWIN, 128, 0, stream>>>(xbuf, out_seeds, normf_g, normf_b,
                                   p_h1, h1_b, p_h2, h2_b, h3_w, h3_b, out_labels);
}
